// SchNetInteraction_88880053223521
// MI455X (gfx1250) — compile-verified, matrix operand order checked by reading, not run on hardware
//
#include <hip/hip_runtime.h>
#include <hip/hip_bf16.h>

typedef __attribute__((ext_vector_type(16))) _Float16 v16h;
typedef __attribute__((ext_vector_type(8)))  float    v8f;

#define DEVI __device__ __forceinline__

// D = A(16x32 f16) x B(32x16 f16) + C (16x16 f32)
DEVI v8f wmma_16x16x32(v16h a, v16h b, v8f c) {
  return __builtin_amdgcn_wmma_f32_16x16x32_f16(
      /*neg_a=*/false, a, /*neg_b=*/false, b,
      /*c_mod=*/(short)0, c, /*reuse_a=*/false, /*reuse_b=*/false);
}

// shifted softplus: softplus(v) - log(2), in log2/exp2 domain with raw HW ops.
// log2 argument is in [1,2] -> no denorm/inf fixups needed.
DEVI float sspf(float v) {
  const float LOG2E = 1.4426950408889634f;
  const float LN2   = 0.6931471805599453f;
  float e = __builtin_amdgcn_exp2f(-fabsf(v) * LOG2E);   // exp(-|v|)
  float l = __builtin_amdgcn_logf(1.0f + e);             // log2(1+exp(-|v|))
  return fmaxf(v, 0.0f) - LN2 + l * LN2;
}

DEVI void put4(v16h& v, int base, float4 f) {
  v[base + 0] = (_Float16)f.x;
  v[base + 1] = (_Float16)f.y;
  v[base + 2] = (_Float16)f.z;
  v[base + 3] = (_Float16)f.w;
}

// ---------------------------------------------------------------------------
// Kernel 1: y[b,n,f] = sum_a x[b,n,a] * w_in2f[f,a]   (tiny: 33 MFLOP)
// ---------------------------------------------------------------------------
__global__ void __launch_bounds__(64) schnet_in2f(
    const float* __restrict__ x, const float* __restrict__ w, float* __restrict__ y) {
  constexpr int AA = 128, F = 64;
  __shared__ float xs[AA];
  const int bn = blockIdx.x;
  const float* xrow = x + (size_t)bn * AA;
  for (int a = threadIdx.x; a < AA; a += 64) xs[a] = xrow[a];
  __syncthreads();
  const int f = threadIdx.x;
  const float* wr = w + (size_t)f * AA;
  float acc = 0.0f;
#pragma unroll 8
  for (int a = 0; a < AA; ++a) acc += xs[a] * wr[a];
  y[(size_t)bn * F + f] = acc;
}

// ---------------------------------------------------------------------------
// Kernel 2: fused filter-network + cfconv + output MLP.
// One workgroup (4 wave32) per (b,i). Per j-tile of 16 neighbors:
//   GEMM1 (transposed): C1[f,j] = w_f1 x f_ij^T + b_f1 (bias in C)   4 WMMA
//   ssp, pack f16 in-lane -> B-operand frags for GEMM2 (no LDS)
//   GEMM2 (transposed): D[f',j] = w_f2 x W1 + b_f2 (bias in C)       8 WMMA
//   acc[f'] += (rcut[i,j] * y[j,f']) * D[f',j]
// Then LDS reduce over j-lanes/waves and run the 64->128->128 output MLP.
// ---------------------------------------------------------------------------
__global__ void __launch_bounds__(128) schnet_main(
    const float* __restrict__ fij,  const float* __restrict__ rcut,
    const float* __restrict__ w_f1, const float* __restrict__ b_f1,
    const float* __restrict__ w_f2, const float* __restrict__ b_f2,
    const float* __restrict__ w_o1, const float* __restrict__ b_o1,
    const float* __restrict__ w_o2, const float* __restrict__ b_o2,
    const float* __restrict__ y,    float* __restrict__ out) {
  constexpr int N = 1024, F = 64, R = 20, AA = 128;

  __shared__ float agg[F];
  __shared__ float hbuf[AA];

  const int tid  = threadIdx.x;
  const int lane = tid & 31;
  const int wave = tid >> 5;
  const int l16  = lane & 15;
  const int hi   = lane >> 4;        // K-half select (0: lanes 0-15, 1: lanes 16-31)
  const int hi8  = hi * 8;
  const int bi   = blockIdx.x;       // b*N + i
  const int b    = bi >> 10;

  for (int f = tid; f < F; f += 128) agg[f] = 0.0f;
  __syncthreads();

  // ---- preload w_f1 A-fragments: tile t covers f in [16t,16t+16), K=r (20 padded to 32)
  v16h A1[4];
#pragma unroll
  for (int t = 0; t < 4; ++t) {
    const float* rr = w_f1 + (size_t)(t * 16 + l16) * R;
    const float* p  = rr + (hi ? 8 : 0);
    float4 a0 = *(const float4*)(p);
    float4 a1 = *(const float4*)(p + 4);
    float4 a2 = *(const float4*)(rr + 16);           // r=16..19
    if (hi) a2 = make_float4(0.f, 0.f, 0.f, 0.f);    // hi half K=24..31 -> pad
    put4(A1[t], 0, a0);
    put4(A1[t], 4, a1);
    put4(A1[t], 8, a2);
    put4(A1[t], 12, make_float4(0.f, 0.f, 0.f, 0.f)); // r=20..23 pad
  }

  // ---- preload w_f2 A-fragments: tile t = f' range, chunk c = K(f) in [32c,32c+32)
  v16h A2[4][2];
#pragma unroll
  for (int t = 0; t < 4; ++t) {
#pragma unroll
    for (int c = 0; c < 2; ++c) {
      const float* rr = w_f2 + (size_t)(t * 16 + l16) * F + c * 32 + (hi ? 8 : 0);
      put4(A2[t][c], 0,  *(const float4*)(rr));
      put4(A2[t][c], 4,  *(const float4*)(rr + 4));
      put4(A2[t][c], 8,  *(const float4*)(rr + 16));
      put4(A2[t][c], 12, *(const float4*)(rr + 20));
    }
  }

  // ---- bias fragments in C/D register layout: element r of tile t -> f = 16t + 8*hi + r
  // Used directly as the WMMA C operand (accumulator init) -> zero VALU bias cost.
  v8f bias1v[4], bias2v[4];
#pragma unroll
  for (int t = 0; t < 4; ++t)
#pragma unroll
    for (int r = 0; r < 8; ++r) {
      bias1v[t][r] = b_f1[t * 16 + hi8 + r];
      bias2v[t][r] = b_f2[t * 16 + hi8 + r];
    }

  float acc[4][8];
#pragma unroll
  for (int t = 0; t < 4; ++t)
#pragma unroll
    for (int r = 0; r < 8; ++r) acc[t][r] = 0.0f;

  const float* fij_base = fij + (size_t)bi * N * R;   // [1024][20]
  const float* rcut_row = rcut + (size_t)bi * N;
  const float* ybase    = y + (size_t)b * N * F;

  // 64 j-tiles of 16, strided across the 4 waves
  for (int jt = wave; jt < N / 16; jt += 4) {
    const int j = jt * 16 + l16;     // this lane's neighbor row

    // prefetch next tile's row for this lane
    if (jt + 4 < N / 16)
      __builtin_prefetch(fij_base + (size_t)(j + 64) * R, 0, 3);

    // ---- B-operand fragment from f_ij row j (K=r, N=j), f32 -> f16
    v16h Bf;
    {
      const float* rr = fij_base + (size_t)j * R;
      const float* p  = rr + (hi ? 8 : 0);
      float4 a0 = *(const float4*)(p);
      float4 a1 = *(const float4*)(p + 4);
      float4 a2 = *(const float4*)(rr + 16);
      if (hi) a2 = make_float4(0.f, 0.f, 0.f, 0.f);
      put4(Bf, 0, a0);
      put4(Bf, 4, a1);
      put4(Bf, 8, a2);
      put4(Bf, 12, make_float4(0.f, 0.f, 0.f, 0.f));
    }

    // ---- GEMM1: C1[t] = w_f1 tile x f_ij^T + b_f1   (D[m=f][n=j])
    v8f C1[4];
#pragma unroll
    for (int t = 0; t < 4; ++t)
      C1[t] = wmma_16x16x32(A1[t], Bf, bias1v[t]);

    // ---- ssp, pack in-lane into GEMM2 B-operand frags (K=f, N=j)
    // C-tile (lane=j, rows f=16t+8*hi+r) matches B-frag lo/hi K-half split exactly.
    v16h Bw0, Bw1;
#pragma unroll
    for (int e = 0; e < 8; ++e) {
      Bw0[e]     = (_Float16)sspf(C1[0][e]);
      Bw0[e + 8] = (_Float16)sspf(C1[1][e]);
      Bw1[e]     = (_Float16)sspf(C1[2][e]);
      Bw1[e + 8] = (_Float16)sspf(C1[3][e]);
    }

    const float rc = rcut_row[j];
    const float* yrow = ybase + (size_t)j * F;

    // ---- GEMM2 + epilogue: D = w_f2 x W1 + b_f2; acc += (rc*y) * D
#pragma unroll
    for (int t = 0; t < 4; ++t) {
      v8f D = wmma_16x16x32(A2[t][0], Bw0, bias2v[t]);
      D     = wmma_16x16x32(A2[t][1], Bw1, D);
      const float* yr = yrow + t * 16 + hi8;
      float4 y0 = *(const float4*)(yr);
      float4 y1 = *(const float4*)(yr + 4);
      acc[t][0] += (rc * y0.x) * D[0];
      acc[t][1] += (rc * y0.y) * D[1];
      acc[t][2] += (rc * y0.z) * D[2];
      acc[t][3] += (rc * y0.w) * D[3];
      acc[t][4] += (rc * y1.x) * D[4];
      acc[t][5] += (rc * y1.y) * D[5];
      acc[t][6] += (rc * y1.z) * D[6];
      acc[t][7] += (rc * y1.w) * D[7];
    }
  }

  // ---- reduce partial sums over j-lanes and waves (ds_add_f32)
#pragma unroll
  for (int t = 0; t < 4; ++t)
#pragma unroll
    for (int r = 0; r < 8; ++r)
      atomicAdd(&agg[t * 16 + hi8 + r], acc[t][r]);
  __syncthreads();

  // ---- output MLP: h = ssp(agg @ w_o1^T + b_o1); out = h @ w_o2^T + b_o2
  {
    const int a = tid;                       // 0..127
    float s = b_o1[a];
    const float* wr = w_o1 + (size_t)a * F;
#pragma unroll 8
    for (int f = 0; f < F; ++f) s += agg[f] * wr[f];
    hbuf[a] = sspf(s);
  }
  __syncthreads();
  {
    const int c = tid;                       // 0..127
    float s = b_o2[c];
    const float* wr = w_o2 + (size_t)c * AA;
#pragma unroll 8
    for (int a2 = 0; a2 < AA; ++a2) s += hbuf[a2] * wr[a2];
    out[(size_t)bi * AA + c] = s;
  }
}

// ---------------------------------------------------------------------------
extern "C" void kernel_launch(void* const* d_in, const int* in_sizes, int n_in,
                              void* d_out, int out_size, void* d_ws, size_t ws_size,
                              hipStream_t stream) {
  const float* x      = (const float*)d_in[0];
  const float* fij    = (const float*)d_in[1];
  const float* rcut   = (const float*)d_in[2];
  const float* w_in2f = (const float*)d_in[3];
  const float* w_f1   = (const float*)d_in[4];
  const float* b_f1   = (const float*)d_in[5];
  const float* w_f2   = (const float*)d_in[6];
  const float* b_f2   = (const float*)d_in[7];
  const float* w_o1   = (const float*)d_in[8];
  const float* b_o1   = (const float*)d_in[9];
  const float* w_o2   = (const float*)d_in[10];
  const float* b_o2   = (const float*)d_in[11];
  float* out = (float*)d_out;
  float* y   = (float*)d_ws;   // [B*N, 64] = 512 KB scratch

  constexpr int BN = 2 * 1024;
  schnet_in2f<<<BN, 64, 0, stream>>>(x, w_in2f, y);
  schnet_main<<<BN, 128, 0, stream>>>(fij, rcut, w_f1, b_f1, w_f2, b_f2,
                                      w_o1, b_o1, w_o2, b_o2, y, out);
}
